// SlotMixerDecoder_52003464020088
// MI455X (gfx1250) — matrix operand order repaired and checked by reading, not compiled, not run on hardware
//
#include <hip/hip_runtime.h>
#include <cstdint>
#include <cstddef>

typedef _Float16 h16;
typedef __attribute__((ext_vector_type(4)))  _Float16 v4h;
typedef __attribute__((ext_vector_type(8)))  _Float16 v8h;
typedef __attribute__((ext_vector_type(16))) _Float16 v16h;
typedef __attribute__((ext_vector_type(8)))  float    v8f;

// ---------------- problem constants (from reference) ----------------
constexpr int BATCH = 4;
constexpr int RAYS  = 8192;
constexpr int NTOT  = BATCH * RAYS;   // 32768 rows, multiple of 128
constexpr int SLOTS = 8;
constexpr int ZDIM  = 1536;
constexpr int NHEAD = 12;
constexpr int DHEAD = 128;
constexpr int IDIM  = 180;
constexpr int IPAD  = 192;            // padded row stride for 180-wide acts
constexpr int MLPD  = 3072;
constexpr int INNER = 1536;           // NHEAD*DHEAD
constexpr int RKIN  = ZDIM + IDIM;    // 1716 render-MLP input
constexpr int RKPAD = 1728;
constexpr float SCALE_ATTN = 0.08838834764831845f;  // 128^-0.5
constexpr float SCALE_MIX  = 0.025515518153991442f; // 1536^-0.5

// ---------------- tiny elementwise / helper kernels ----------------
__global__ void k_cvt(const float* __restrict__ in, h16* __restrict__ out, int n) {
  int i = blockIdx.x * blockDim.x + threadIdx.x;
  if (i < n) out[i] = (h16)in[i];
}

// convert weight [K,N] f32 -> [Kpad,N] f16 with zero-filled pad rows
__global__ void k_cvt_pad(const float* __restrict__ in, h16* __restrict__ out,
                          int K, int N, int Kpad) {
  int i = blockIdx.x * blockDim.x + threadIdx.x;
  if (i >= Kpad * N) return;
  int r = i / N;
  out[i] = (r < K) ? (h16)in[i] : (h16)0.f;
}

// positional encoding -> queries f16 [NTOT, IPAD]
__global__ void k_pe(const float* __restrict__ cam, const float* __restrict__ ray,
                     h16* __restrict__ q16) {
  int g = blockIdx.x * blockDim.x + threadIdx.x;
  if (g >= NTOT) return;
  h16* out = q16 + (size_t)g * IPAD;
  const float* srcs[2] = { cam + (size_t)g * 3, ray + (size_t)g * 3 };
  for (int half = 0; half < 2; ++half) {
    const float* c = srcs[half];
    int base = half * 90;
    for (int d = 0; d < 3; ++d) {
      float x = c[d] * 3.14159265358979323846f;
      for (int o = 0; o < 15; ++o) {
        out[base + d * 15 + o]      = (h16)__sinf(x);
        out[base + 45 + d * 15 + o] = (h16)__cosf(x);
        x *= 2.0f;
      }
    }
  }
  for (int j = 2 * 90; j < IPAD; ++j) out[j] = (h16)0.f;
}

// LayerNorm over 180 features: f32 in [NTOT,IPAD] -> f16 out [NTOT,IPAD]
__global__ void k_ln(const float* __restrict__ x, const float* __restrict__ g,
                     const float* __restrict__ b, h16* __restrict__ o) {
  int r = blockIdx.x * blockDim.x + threadIdx.x;
  if (r >= NTOT) return;
  const float* xr = x + (size_t)r * IPAD;
  float m = 0.f;
  for (int j = 0; j < IDIM; ++j) m += xr[j];
  m *= (1.f / IDIM);
  float v = 0.f;
  for (int j = 0; j < IDIM; ++j) { float d = xr[j] - m; v += d * d; }
  v *= (1.f / IDIM);
  float rstd = rsqrtf(v + 1e-5f);
  h16* orow = o + (size_t)r * IPAD;
  for (int j = 0; j < IDIM; ++j) orow[j] = (h16)((xr[j] - m) * rstd * g[j] + b[j]);
  for (int j = IDIM; j < IPAD; ++j) orow[j] = (h16)0.f;
}

// kv = slot_latents @ Wkv (32 rows only) -> k16,v16 [BATCH*SLOTS, INNER] f16
__global__ void k_kv(const float* __restrict__ slot, const float* __restrict__ Wkv,
                     h16* __restrict__ k16, h16* __restrict__ v16) {
  int idx = blockIdx.x * blockDim.x + threadIdx.x;
  if (idx >= BATCH * SLOTS * 2 * INNER) return;
  int row = idx / (2 * INNER), oc = idx % (2 * INNER);
  const float* s = slot + (size_t)row * ZDIM;
  float acc = 0.f;
  for (int d = 0; d < ZDIM; ++d) acc = fmaf(s[d], Wkv[(size_t)d * (2 * INNER) + oc], acc);
  if (oc < INNER) k16[(size_t)row * INNER + oc] = (h16)acc;
  else            v16[(size_t)row * INNER + (oc - INNER)] = (h16)acc;
}

// mk = slot_latents @ Wmk  -> f32 [BATCH*SLOTS, INNER]
__global__ void k_mk(const float* __restrict__ slot, const float* __restrict__ Wmk,
                     float* __restrict__ mk) {
  int idx = blockIdx.x * blockDim.x + threadIdx.x;
  if (idx >= BATCH * SLOTS * INNER) return;
  int row = idx / INNER, oc = idx % INNER;
  const float* s = slot + (size_t)row * ZDIM;
  float acc = 0.f;
  for (int d = 0; d < ZDIM; ++d) acc = fmaf(s[d], Wmk[(size_t)d * INNER + oc], acc);
  mk[(size_t)row * INNER + oc] = acc;
}

// 8-slot cross attention, one thread per (ray, head)
__global__ void k_attn(const h16* __restrict__ q16, const h16* __restrict__ k16,
                       const h16* __restrict__ v16, h16* __restrict__ o16) {
  int idx = blockIdx.x * blockDim.x + threadIdx.x;
  if (idx >= NTOT * NHEAD) return;
  int g = idx / NHEAD, h = idx % NHEAD;
  int b = g / RAYS;
  const h16* q = q16 + (size_t)g * INNER + h * DHEAD;
  float dots[SLOTS];
  float mx = -1e30f;
  for (int s = 0; s < SLOTS; ++s) {
    const h16* kp = k16 + ((size_t)(b * SLOTS + s) * INNER + h * DHEAD);
    float a = 0.f;
    for (int d = 0; d < DHEAD; ++d) a = fmaf((float)q[d], (float)kp[d], a);
    a *= SCALE_ATTN;
    dots[s] = a;
    mx = fmaxf(mx, a);
  }
  float z = 0.f;
  for (int s = 0; s < SLOTS; ++s) { dots[s] = __expf(dots[s] - mx); z += dots[s]; }
  float rz = 1.f / z;
  for (int s = 0; s < SLOTS; ++s) dots[s] *= rz;
  h16* orow = o16 + (size_t)g * INNER + h * DHEAD;
  for (int d = 0; d < DHEAD; ++d) {
    float a = 0.f;
    for (int s = 0; s < SLOTS; ++s)
      a = fmaf(dots[s], (float)v16[(size_t)(b * SLOTS + s) * INNER + h * DHEAD + d], a);
    orow[d] = (h16)a;
  }
}

// mixing softmax + slot mix + concat(queries): writes w (output) and render input
__global__ void k_mix(const h16* __restrict__ mq16, const float* __restrict__ mk,
                      const float* __restrict__ slot, const h16* __restrict__ q16,
                      float* __restrict__ w_out, h16* __restrict__ rin) {
  int g = blockIdx.x * blockDim.x + threadIdx.x;
  if (g >= NTOT) return;
  int b = g / RAYS;
  const h16* mq = mq16 + (size_t)g * INNER;
  float logit[SLOTS];
  float mx = -1e30f;
  for (int s = 0; s < SLOTS; ++s) {
    const float* mkr = mk + (size_t)(b * SLOTS + s) * INNER;
    float a = 0.f;
    for (int d = 0; d < INNER; ++d) a = fmaf((float)mq[d], mkr[d], a);
    a *= SCALE_MIX;
    logit[s] = a;
    mx = fmaxf(mx, a);
  }
  float z = 0.f;
  for (int s = 0; s < SLOTS; ++s) { logit[s] = __expf(logit[s] - mx); z += logit[s]; }
  float rz = 1.f / z;
  for (int s = 0; s < SLOTS; ++s) {
    logit[s] *= rz;
    w_out[(size_t)g * SLOTS + s] = logit[s];
  }
  h16* rrow = rin + (size_t)g * RKPAD;
  for (int d = 0; d < ZDIM; ++d) {
    float a = 0.f;
    for (int s = 0; s < SLOTS; ++s)
      a = fmaf(logit[s], slot[(size_t)(b * SLOTS + s) * ZDIM + d], a);
    rrow[d] = (h16)a;
  }
  const h16* qrow = q16 + (size_t)g * IPAD;
  for (int j = 0; j < IDIM; ++j) rrow[ZDIM + j] = qrow[j];
  for (int j = RKIN; j < RKPAD; ++j) rrow[j] = (h16)0.f;
}

// final 1536 -> 3 projection, thread per (ray, channel)
__global__ void k_final(const h16* __restrict__ hbuf, const float* __restrict__ w5,
                        const float* __restrict__ b5, float* __restrict__ pix) {
  int idx = blockIdx.x * blockDim.x + threadIdx.x;
  if (idx >= NTOT * 3) return;
  int g = idx / 3, c = idx % 3;
  const h16* hr = hbuf + (size_t)g * INNER;
  float acc = b5[c];
  for (int d = 0; d < INNER; ++d) acc = fmaf((float)hr[d], w5[(size_t)d * 3 + c], acc);
  pix[(size_t)g * 3 + c] = acc;
}

// ---------------- WMMA GEMM: C = act(A[M,Kpad] @ W[Kpad,N] + bias) ----------------
// Kpad multiple of 32; weight pad rows are zero so no K guards needed.
// Block tile 128x64, 8 waves (wave32), wave tile 32x32 = 2x2 v_wmma_f32_16x16x32_f16.
// Double-buffered LDS, register-staged global loads overlapped with WMMA,
// last iteration peeled so the steady-state loop is branch-free.
// ACT: 0=none 1=relu 2=gelu(exact) 3=leaky(0.01)
// MODE: 0 store f16 to outH, 1 store f32 to outF, 2 add into f32 outF (residual)
template<int ACT, int MODE>
__global__ __launch_bounds__(256)
void k_gemm(const h16* __restrict__ A, int lda, const h16* __restrict__ Bw, int ldb,
            const float* __restrict__ bias, h16* __restrict__ outH,
            float* __restrict__ outF, int ldc, int N, int Kpad) {
  constexpr int BM = 128, BN = 64, BK = 32;
  __shared__ h16 sA[2][BM][BK + 8];   // 80B row stride: 16B aligned
  __shared__ h16 sB[2][BK][BN + 8];   // 144B row stride: 16B aligned
  const int tid  = threadIdx.x;
  const int m0   = blockIdx.x * BM;
  const int n0   = blockIdx.y * BN;
  const int wave = tid >> 5, lane = tid & 31;
  const int wm = wave >> 1, wn = wave & 1;

  // cooperative staging coordinates (unguarded; memory is padded)
  const int arow = tid >> 1;          // 0..127
  const int akb  = (tid & 1) * 16;    // 0 / 16
  const int bkr  = tid >> 3;          // 0..31
  const int bnb  = (tid & 7) * 8;     // 0..56
  const h16* aptr = A + (size_t)(m0 + arow) * lda + akb;
  const h16* bptr = Bw + (size_t)bkr * ldb + n0 + bnb;

  v8h ra0, ra1;   // staged A: 16 halves
  v4h rb0, rb1;   // staged B: 8 halves

  auto gload = [&](int k0) {
    const v8h* ap = (const v8h*)(aptr + k0);                // 32B-aligned
    ra0 = ap[0]; ra1 = ap[1];
    const v4h* bp = (const v4h*)(bptr + (size_t)k0 * ldb);  // 8B-aligned
    rb0 = bp[0]; rb1 = bp[1];
  };
  auto sstore = [&](int buf) {
    *(v8h*)&sA[buf][arow][akb]     = ra0;
    *(v8h*)&sA[buf][arow][akb + 8] = ra1;
    *(v4h*)&sB[buf][bkr][bnb]      = rb0;
    *(v4h*)&sB[buf][bkr][bnb + 4]  = rb1;
  };

  v8f acc[2][2] = {};

  auto compute = [&](int cur) {
    // fragments (documented 16-bit A 16x32 / B 32x16 layouts)
    v16h afrag[2], bfrag[2];
#pragma unroll
    for (int mi = 0; mi < 2; ++mi) {
      int row   = wm * 32 + mi * 16 + (lane & 15);
      int khalf = (lane >> 4) * 8;
#pragma unroll
      for (int v = 0; v < 8; ++v) {
        int kk = (v < 4) ? (khalf + 2 * v) : (16 + khalf + 2 * (v - 4));
        afrag[mi][2 * v]     = sA[cur][row][kk];
        afrag[mi][2 * v + 1] = sA[cur][row][kk + 1];
      }
    }
#pragma unroll
    for (int ni = 0; ni < 2; ++ni) {
      int nb = wn * 32 + ni * 16;
#pragma unroll
      for (int j = 0; j < 16; ++j) bfrag[ni][j] = sB[cur][(j < 8) ? ((lane >> 4) * 8 + j) : (16 + (lane >> 4) * 8 + (j - 8))][nb + (lane & 15)];
    }
#pragma unroll
    for (int mi = 0; mi < 2; ++mi)
#pragma unroll
      for (int ni = 0; ni < 2; ++ni)
        acc[mi][ni] = __builtin_amdgcn_wmma_f32_16x16x32_f16(
            false, afrag[mi], false, bfrag[ni], (short)0, acc[mi][ni], false, false);
  };

  const int T = Kpad / BK;   // >= 1
  gload(0);
  sstore(0);
  for (int t = 0; t < T - 1; ++t) {    // branch-free steady state
    gload((t + 1) * BK);               // issue next tile's loads early
    __syncthreads();                   // tile t visible in LDS
    compute(t & 1);                    // WMMA while loads are in flight
    sstore((t + 1) & 1);               // drain staged regs into other buffer
  }
  __syncthreads();
  compute((T - 1) & 1);                // peeled last tile

  // ---- epilogue: C layout lane<16 -> M=r, lane>=16 -> M=r+8; col = lane&15 ----
#pragma unroll
  for (int mi = 0; mi < 2; ++mi) {
#pragma unroll
    for (int ni = 0; ni < 2; ++ni) {
      int col = n0 + wn * 32 + ni * 16 + (lane & 15);
      if (col >= N) continue;
      float bv = bias ? bias[col] : 0.f;
#pragma unroll
      for (int r = 0; r < 8; ++r) {
        int row = m0 + wm * 32 + mi * 16 + ((lane < 16) ? r : r + 8);
        float v = acc[mi][ni][r] + bv;
        if constexpr (ACT == 1) v = fmaxf(v, 0.f);
        if constexpr (ACT == 2) v = 0.5f * v * (1.f + erff(v * 0.70710678118654752f));
        if constexpr (ACT == 3) v = (v > 0.f) ? v : 0.01f * v;
        size_t o = (size_t)row * ldc + col;
        if constexpr (MODE == 0) outH[o] = (h16)v;
        if constexpr (MODE == 1) outF[o] = v;
        if constexpr (MODE == 2) outF[o] += v;
      }
    }
  }
}

// ---------------- host orchestration ----------------
static inline void* ws_take(char* base, size_t& off, size_t bytes) {
  void* p = base + off;
  off = (off + bytes + 255) & ~(size_t)255;
  return p;
}

template<int ACT, int MODE>
static void launch_gemm(const h16* A, int lda, const h16* Bw, int ldb,
                        const float* bias, h16* oh, float* of, int ldc,
                        int N, int Kpad, hipStream_t stream) {
  dim3 grid(NTOT / 128, (N + 63) / 64);
  k_gemm<ACT, MODE><<<grid, 256, 0, stream>>>(A, lda, Bw, ldb, bias, oh, of, ldc, N, Kpad);
}

extern "C" void kernel_launch(void* const* d_in, const int* in_sizes, int n_in,
                              void* d_out, int out_size, void* d_ws, size_t ws_size,
                              hipStream_t stream) {
  (void)in_sizes; (void)n_in; (void)out_size; (void)ws_size;
  // inputs per setup_inputs() order
  const float* slot   = (const float*)d_in[0];
  const float* cam    = (const float*)d_in[1];
  const float* rays   = (const float*)d_in[2];
  const float* w_in1  = (const float*)d_in[3];
  const float* b_in1  = (const float*)d_in[4];
  const float* w_in2  = (const float*)d_in[5];
  const float* b_in2  = (const float*)d_in[6];
  const float* ln1_g  = (const float*)d_in[7];
  const float* ln1_b  = (const float*)d_in[8];
  const float* Wq     = (const float*)d_in[9];
  const float* Wkv    = (const float*)d_in[10];
  const float* Wo     = (const float*)d_in[11];
  const float* bo     = (const float*)d_in[12];
  const float* ln2_g  = (const float*)d_in[13];
  const float* ln2_b  = (const float*)d_in[14];
  const float* Wf1    = (const float*)d_in[15];
  const float* bf1    = (const float*)d_in[16];
  const float* Wf2    = (const float*)d_in[17];
  const float* bf2    = (const float*)d_in[18];
  const float* Wmq    = (const float*)d_in[19];
  const float* Wmk    = (const float*)d_in[20];
  const float* rw1    = (const float*)d_in[21];
  const float* rb1    = (const float*)d_in[22];
  const float* rw2    = (const float*)d_in[23];
  const float* rb2    = (const float*)d_in[24];
  const float* rw3    = (const float*)d_in[25];
  const float* rb3    = (const float*)d_in[26];
  const float* rw4    = (const float*)d_in[27];
  const float* rb4    = (const float*)d_in[28];
  const float* rw5    = (const float*)d_in[29];
  const float* rb5    = (const float*)d_in[30];

  float* pixels = (float*)d_out;                       // [NTOT,3]
  float* w_out  = (float*)d_out + (size_t)NTOT * 3;    // [NTOT,8]

  char* ws = (char*)d_ws;
  size_t off = 0;
  // f16 weight pool: Kpad rows (+64 element slack for last-column-block overreads)
  h16* w_in1_16 = (h16*)ws_take(ws, off, sizeof(h16) * (192 * 360 + 64));
  h16* w_in2_16 = (h16*)ws_take(ws, off, sizeof(h16) * (384 * 180 + 64));
  h16* Wq16     = (h16*)ws_take(ws, off, sizeof(h16) * (2 * 192 * 1536 + 64));
  h16* Wo16     = (h16*)ws_take(ws, off, sizeof(h16) * (2 * 1536 * 180 + 64));
  h16* Wf1_16   = (h16*)ws_take(ws, off, sizeof(h16) * (2 * 192 * 3072 + 64));
  h16* Wf2_16   = (h16*)ws_take(ws, off, sizeof(h16) * (2 * 3072 * 180 + 64));
  h16* Wmq16    = (h16*)ws_take(ws, off, sizeof(h16) * (192 * 1536 + 64));
  h16* rw1_16   = (h16*)ws_take(ws, off, sizeof(h16) * (1728 * 1536 + 64));
  h16* rw2_16   = (h16*)ws_take(ws, off, sizeof(h16) * (1536 * 1536 + 64));
  h16* rw3_16   = (h16*)ws_take(ws, off, sizeof(h16) * (1536 * 1536 + 64));
  h16* rw4_16   = (h16*)ws_take(ws, off, sizeof(h16) * (1536 * 1536 + 64));
  // activations
  h16*   Q16 = (h16*)ws_take(ws, off, sizeof(h16) * (size_t)NTOT * IPAD);
  float* X32 = (float*)ws_take(ws, off, sizeof(float) * (size_t)NTOT * IPAD);
  h16*   H16 = (h16*)ws_take(ws, off, sizeof(h16) * (size_t)NTOT * IPAD);
  h16*   BA  = (h16*)ws_take(ws, off, sizeof(h16) * (size_t)NTOT * INNER);
  h16*   BB  = (h16*)ws_take(ws, off, sizeof(h16) * (size_t)NTOT * INNER);
  h16*   B3  = (h16*)ws_take(ws, off, sizeof(h16) * (size_t)NTOT * MLPD);  // also [NTOT,RKPAD]/[NTOT,384]
  h16*   K16 = (h16*)ws_take(ws, off, sizeof(h16) * BATCH * SLOTS * INNER);
  h16*   V16 = (h16*)ws_take(ws, off, sizeof(h16) * BATCH * SLOTS * INNER);
  float* MK32 = (float*)ws_take(ws, off, sizeof(float) * BATCH * SLOTS * INNER);

  auto cvtpad = [&](const float* src, h16* dst, int K, int N, int Kpad) {
    int n = Kpad * N;
    k_cvt_pad<<<(n + 255) / 256, 256, 0, stream>>>(src, dst, K, N, Kpad);
  };

  // ---- weight conversions with K zero-padding (every call; deterministic) ----
  cvtpad(w_in1, w_in1_16, 180, 360, 192);
  cvtpad(w_in2, w_in2_16, 360, 180, 384);
  for (int i = 0; i < 2; ++i) {
    cvtpad(Wq  + (size_t)i * 180 * 1536, Wq16   + (size_t)i * 192 * 1536, 180, 1536, 192);
    cvtpad(Wo  + (size_t)i * 1536 * 180, Wo16   + (size_t)i * 1536 * 180, 1536, 180, 1536);
    cvtpad(Wf1 + (size_t)i * 180 * 3072, Wf1_16 + (size_t)i * 192 * 3072, 180, 3072, 192);
    cvtpad(Wf2 + (size_t)i * 3072 * 180, Wf2_16 + (size_t)i * 3072 * 180, 3072, 180, 3072);
  }
  cvtpad(Wmq, Wmq16, 180, 1536, 192);
  cvtpad(rw1, rw1_16, RKIN, 1536, RKPAD);
  cvtpad(rw2, rw2_16, 1536, 1536, 1536);
  cvtpad(rw3, rw3_16, 1536, 1536, 1536);
  cvtpad(rw4, rw4_16, 1536, 1536, 1536);

  // ---- ray encoder + input MLP ----
  k_pe<<<(NTOT + 255) / 256, 256, 0, stream>>>(cam, rays, Q16);
  launch_gemm<1,0>(Q16, IPAD, w_in1_16, 360, b_in1, B3, nullptr, 384, 360, 192, stream);
  launch_gemm<0,1>(B3, 384, w_in2_16, IDIM, b_in2, nullptr, X32, IPAD, IDIM, 384, stream);

  // ---- 2 cross-attention transformer blocks ----
  for (int i = 0; i < 2; ++i) {
    k_ln<<<(NTOT + 255) / 256, 256, 0, stream>>>(X32, ln1_g + i * IDIM, ln1_b + i * IDIM, H16);
    launch_gemm<0,0>(H16, IPAD, Wq16 + (size_t)i * 192 * 1536, INNER, nullptr,
                     BA, nullptr, INNER, INNER, 192, stream);
    k_kv<<<(BATCH * SLOTS * 2 * INNER + 255) / 256, 256, 0, stream>>>(
        slot, Wkv + (size_t)i * ZDIM * 2 * INNER, K16, V16);
    k_attn<<<(NTOT * NHEAD + 255) / 256, 256, 0, stream>>>(BA, K16, V16, BB);
    launch_gemm<0,2>(BB, INNER, Wo16 + (size_t)i * 1536 * 180, IDIM, bo + i * IDIM,
                     nullptr, X32, IPAD, IDIM, 1536, stream);
    k_ln<<<(NTOT + 255) / 256, 256, 0, stream>>>(X32, ln2_g + i * IDIM, ln2_b + i * IDIM, H16);
    launch_gemm<2,0>(H16, IPAD, Wf1_16 + (size_t)i * 192 * 3072, MLPD, bf1 + i * MLPD,
                     B3, nullptr, MLPD, MLPD, 192, stream);
    launch_gemm<0,2>(B3, MLPD, Wf2_16 + (size_t)i * 3072 * 180, IDIM, bf2 + i * IDIM,
                     nullptr, X32, IPAD, IDIM, 3072, stream);
  }

  // ---- mixing block ----
  k_cvt<<<(NTOT * IPAD + 255) / 256, 256, 0, stream>>>(X32, H16, NTOT * IPAD);
  launch_gemm<0,0>(H16, IPAD, Wmq16, INNER, nullptr, BA, nullptr, INNER, INNER, 192, stream);
  k_mk<<<(BATCH * SLOTS * INNER + 255) / 256, 256, 0, stream>>>(slot, Wmk, MK32);
  k_mix<<<(NTOT + 255) / 256, 256, 0, stream>>>(BA, MK32, slot, Q16, w_out, B3);

  // ---- render MLP ----
  launch_gemm<3,0>(B3, RKPAD, rw1_16, 1536, rb1, BA, nullptr, 1536, 1536, RKPAD, stream);
  launch_gemm<3,0>(BA, 1536, rw2_16, 1536, rb2, BB, nullptr, 1536, 1536, 1536, stream);
  launch_gemm<3,0>(BB, 1536, rw3_16, 1536, rb3, BA, nullptr, 1536, 1536, 1536, stream);
  launch_gemm<3,0>(BA, 1536, rw4_16, 1536, rb4, BB, nullptr, 1536, 1536, 1536, stream);
  k_final<<<(NTOT * 3 + 255) / 256, 256, 0, stream>>>(BB, rw5, rb5, pixels);
}
